// BeliefDecoder_49048526520412
// MI455X (gfx1250) — matrix operand order repaired and checked by reading, not compiled, not run on hardware
//
#include <hip/hip_runtime.h>
#include <hip/hip_bf16.h>
#include <stdint.h>

// ---------------------------------------------------------------------------
// Types
// ---------------------------------------------------------------------------
typedef __attribute__((ext_vector_type(16))) __bf16 v16bf;
typedef __attribute__((ext_vector_type(8)))  __bf16 bf16x8;
typedef __attribute__((ext_vector_type(4)))  __bf16 bf16x4;
typedef __attribute__((ext_vector_type(8)))  float  v8f;
typedef __attribute__((ext_vector_type(4)))  float  f32x4;

#define BATCH    65536
#define NC_STEPS 15
#define VOCAB    26
#define EMB      64
#define HID      256
#define KTOT     576          // 256 ctx + 64 prev + 256 h
#define KC_GATE  18           // 576 / 32
#define KC_H     8            // 256 / 32

#define MTILE    32           // batch rows per workgroup (2 waves x 16 rows)
#define XSTR     584          // x_cat row stride (halves), 16B-aligned rows
#define HSTR     264          // h-stage row stride (halves), 16B-aligned rows

// Workspace layout (bytes)
#define WCAT_OFF 0
#define WCAT_BYTES (64 * KC_GATE * 32 * 16 * 2)   // 1,179,648
#define WP_OFF   (WCAT_OFF + WCAT_BYTES)
#define WP_BYTES (16 * KC_H * 32 * 16 * 2)        // 131,072
#define WO_OFF   (WP_OFF + WP_BYTES)
#define WO_BYTES (2 * KC_H * 32 * 16 * 2)         // 16,384

// ---------------------------------------------------------------------------
// Helpers
// ---------------------------------------------------------------------------
__device__ __forceinline__ v8f wmma_bf16(v16bf a, v16bf b, v8f c) {
  // D = A(16x32 bf16) x B(32x16 bf16) + C(16x16 f32)
  return __builtin_amdgcn_wmma_f32_16x16x32_bf16(
      /*neg_a=*/false, a, /*neg_b=*/false, b,
      /*c_mod=*/(short)0, c, /*reuse_a=*/false, /*reuse_b=*/false);
}

// A fragment (16x32 bf16) from an LDS row-major region.
// Layout: lane l -> row M = l%16; halves 0..15 = K k0+16*(l/16)..+15
// (contiguous order, matching the packed B panels).
__device__ __forceinline__ v16bf load_a(const __bf16* base, int stride,
                                        int mbase, int lane, int k0) {
  const int m  = mbase + (lane & 15);
  const int kk = k0 + ((lane >> 4) << 4);
  const __bf16* q = base + m * stride + kk;
  bf16x8 lo = *(const bf16x8*)q;
  bf16x8 hi = *(const bf16x8*)(q + 8);
  return __builtin_shufflevector(lo, hi, 0, 1, 2, 3, 4, 5, 6, 7,
                                 8, 9, 10, 11, 12, 13, 14, 15);
}

// B fragment (32x16 bf16) from a pre-packed panel: one 32B contiguous load.
__device__ __forceinline__ v16bf load_b(const __bf16* pack, int tile, int nkc,
                                        int kc, int lane) {
  return *(const v16bf*)(pack + (size_t)(tile * nkc + kc) * 512 + lane * 16);
}

__device__ __forceinline__ bf16x4 cvt4(f32x4 v) {
  bf16x4 r;
  r.x = (__bf16)v.x; r.y = (__bf16)v.y; r.z = (__bf16)v.z; r.w = (__bf16)v.w;
  return r;
}

// Branch-free fast activations (no OCML exec-mask paths in the WMMA region).
__device__ __forceinline__ float fast_rcp(float x) {
  return __builtin_amdgcn_rcpf(x);          // v_rcp_f32
}

__device__ __forceinline__ float fast_tanh(float x) {
#if __has_builtin(__builtin_amdgcn_tanhf)
  return __builtin_amdgcn_tanhf(x);         // CDNA5 v_tanh_f32
#else
  // tanh(x) = 1 - 2/(e^{2x}+1); saturates correctly at +/-inf, branch-free.
  float t = __expf(2.0f * x);               // v_exp_f32
  return 1.0f - 2.0f * fast_rcp(t + 1.0f);
#endif
}

__device__ __forceinline__ float sigm(float x) {
  return fast_rcp(1.0f + __expf(-x));       // v_exp_f32 + v_rcp_f32
}

__device__ __forceinline__ uint32_t wanghash(uint32_t x) {
  x ^= x >> 16; x *= 0x7feb352dU;
  x ^= x >> 15; x *= 0x846ca68bU;
  x ^= x >> 16;
  return x;
}

// ---------------------------------------------------------------------------
// Prologue: pack weights into bf16 B-fragment-native panels.
// out[((tile*NKC + kc)*32 + lane)*16 + i] = W[k0 + (lane/16)*16 + i][tile*16 + lane%16]
// ---------------------------------------------------------------------------
__global__ void pack_wcat_kernel(const float* __restrict__ Wi,
                                 const float* __restrict__ Wh,
                                 __bf16* __restrict__ out) {
  int idx  = blockIdx.x * 256 + threadIdx.x;      // 64*18*512 total
  int i    = idx & 15;
  int lane = (idx >> 4) & 31;
  int kc   = (idx >> 9) % KC_GATE;
  int t    = (idx >> 9) / KC_GATE;
  int k = kc * 32 + (lane >> 4) * 16 + i;
  int n = t * 16 + (lane & 15);
  float v = (k < 320) ? Wi[(size_t)k * 1024 + n]
                      : Wh[(size_t)(k - 320) * 1024 + n];
  out[idx] = (__bf16)v;
}

__global__ void pack_wp_kernel(const float* __restrict__ Wp,
                               __bf16* __restrict__ out) {
  int idx  = blockIdx.x * 256 + threadIdx.x;      // 16*8*512 total
  int i    = idx & 15;
  int lane = (idx >> 4) & 31;
  int kc   = (idx >> 9) % KC_H;
  int t    = (idx >> 9) / KC_H;
  int k = kc * 32 + (lane >> 4) * 16 + i;
  int n = t * 16 + (lane & 15);
  out[idx] = (__bf16)Wp[(size_t)k * 256 + n];
}

__global__ void pack_wo_kernel(const float* __restrict__ Wo,
                               __bf16* __restrict__ out) {
  int idx  = blockIdx.x * 256 + threadIdx.x;      // 2*8*512 total
  int i    = idx & 15;
  int lane = (idx >> 4) & 31;
  int kc   = (idx >> 9) % KC_H;
  int t    = (idx >> 9) / KC_H;
  int k = kc * 32 + (lane >> 4) * 16 + i;
  int n = t * 16 + (lane & 15);
  float v = (n < VOCAB) ? Wo[(size_t)k * VOCAB + n] : 0.0f;
  out[idx] = (__bf16)v;
}

// ---------------------------------------------------------------------------
// Main persistent decoder kernel: 2 waves x 16 rows = 32 batch rows per block.
// ---------------------------------------------------------------------------
__global__ __launch_bounds__(64) void belief_decoder_main(
    const float* __restrict__ ctx, const float* __restrict__ embed,
    const float* __restrict__ start_embed, const float* __restrict__ bp,
    const float* __restrict__ bh, const float* __restrict__ bo,
    const __bf16* __restrict__ wcat, const __bf16* __restrict__ wp,
    const __bf16* __restrict__ wo, float* __restrict__ logits_out,
    int* __restrict__ samples_out, const int* __restrict__ seed_ptr) {
  __shared__ __bf16 xcat[MTILE * XSTR];   // [ctx 0..255 | prev 256..319 | h 320..575]
  __shared__ __bf16 hstg[MTILE * HSTR];   // h_new staging (bf16)
  __shared__ float  lsc[2][16][32];       // per-wave logits scratch
  __shared__ float  s_bh[1024];
  __shared__ float  s_bp[256];
  __shared__ float  s_bo[32];

  const int tid  = threadIdx.x;
  const int lane = tid & 31;
  const int wave = tid >> 5;
  const int lo16 = lane & 15;
  const int hi   = lane >> 4;
  const int m0   = wave * 16;
  const int rows_base = blockIdx.x * MTILE;
  const int seed = seed_ptr[0];

  // ---- preload biases ----
  for (int i = tid; i < 1024; i += 64) s_bh[i] = bh[i];
  for (int i = tid; i < 256; i += 64)  s_bp[i] = bp[i];
  if (tid < 32) s_bo[tid] = (tid < VOCAB) ? bo[tid] : 0.0f;

  // ---- stage context (fp32 -> bf16) and start_embed into x_cat ----
  {
    const int row = m0 + lo16;
    const size_t gbase = (size_t)(rows_base + row) * HID;
    const int c0 = hi * 128;
    for (int c = c0; c < c0 + 128; c += 4) {
      f32x4 v = *(const f32x4*)&ctx[gbase + c];
      *(bf16x4*)&xcat[row * XSTR + c] = cvt4(v);
    }
    const int e0 = hi * 32;
    for (int c = e0; c < e0 + 32; c += 4) {
      f32x4 v = *(const f32x4*)&start_embed[c];
      *(bf16x4*)&xcat[row * XSTR + 256 + c] = cvt4(v);
    }
  }
  __syncthreads();

  // ---- initial state: h0 = tanh(ctx @ Wp + bp), c0 = 0 ----
  v8f cfr[16];
#pragma unroll
  for (int nc = 0; nc < 16; ++nc) {
    v8f acc = 0.0f;
    for (int kc = 0; kc < KC_H; ++kc) {
      v16bf a = load_a(xcat, XSTR, m0, lane, kc * 32);  // K over ctx cols only
      v16bf b = load_b(wp, nc, KC_H, kc, lane);
      acc = wmma_bf16(a, b, acc);
    }
    const int ncol = nc * 16 + lo16;
    const float bpv = s_bp[ncol];
#pragma unroll
    for (int j = 0; j < 8; ++j) {
      float hv = fast_tanh(acc[j] + bpv);
      xcat[(m0 + j + 8 * hi) * XSTR + 320 + ncol] = (__bf16)hv;
      cfr[nc][j] = 0.0f;
    }
  }

  // ---- decode steps (all wave-private; no barriers in the loop) ----
#pragma unroll 1
  for (int step = 0; step < NC_STEPS; ++step) {
    // gates = x_cat @ [Wi;Wh] + bh ; LSTM update
#pragma unroll
    for (int nc = 0; nc < 16; ++nc) {
      v8f ai = 0.0f, af = 0.0f, ag = 0.0f, ao = 0.0f;
      const int ti = nc, tf = 16 + nc, tg = 32 + nc, to = 48 + nc;
      for (int kc = 0; kc < KC_GATE; ++kc) {
        v16bf a  = load_a(xcat, XSTR, m0, lane, kc * 32);
        v16bf b0 = load_b(wcat, ti, KC_GATE, kc, lane);
        v16bf b1 = load_b(wcat, tf, KC_GATE, kc, lane);
        v16bf b2 = load_b(wcat, tg, KC_GATE, kc, lane);
        v16bf b3 = load_b(wcat, to, KC_GATE, kc, lane);
        __builtin_prefetch(wcat + (size_t)(ti * KC_GATE + kc + 1) * 512, 0, 0);
        ai = wmma_bf16(a, b0, ai);
        af = wmma_bf16(a, b1, af);
        ag = wmma_bf16(a, b2, ag);
        ao = wmma_bf16(a, b3, ao);
      }
      const int ncol = nc * 16 + lo16;
      const float bi_ = s_bh[ncol];
      const float bf_ = s_bh[256 + ncol];
      const float bg_ = s_bh[512 + ncol];
      const float bo_ = s_bh[768 + ncol];
#pragma unroll
      for (int j = 0; j < 8; ++j) {
        float iv = sigm(ai[j] + bi_);
        float fv = sigm(af[j] + bf_);
        float gv = fast_tanh(ag[j] + bg_);
        float ov = sigm(ao[j] + bo_);
        float cn = fv * cfr[nc][j] + iv * gv;
        cfr[nc][j] = cn;
        float hn = ov * fast_tanh(cn);
        hstg[(m0 + j + 8 * hi) * HSTR + ncol] = (__bf16)hn;
      }
    }

    // logits = h_new @ Wo + bo
#pragma unroll
    for (int nt = 0; nt < 2; ++nt) {
      v8f acc = 0.0f;
      for (int kc = 0; kc < KC_H; ++kc) {
        v16bf a = load_a(hstg, HSTR, m0, lane, kc * 32);
        v16bf b = load_b(wo, nt, KC_H, kc, lane);
        acc = wmma_bf16(a, b, acc);
      }
      const int ncol = nt * 16 + lo16;
      const float bb = s_bo[ncol];
#pragma unroll
      for (int j = 0; j < 8; ++j)
        lsc[wave][j + 8 * hi][ncol] = acc[j] + bb;
    }

    // Gumbel-argmax sampling + embedding gather (one lane per row)
    if (lane < 16) {
      const int m = lo16;
      const int grow = rows_base + m0 + m;
      float best = -1e30f;
      int bidx = 0;
      float* lrow = &lsc[wave][m][0];
      float* gout = &logits_out[((size_t)grow * NC_STEPS + step) * VOCAB];
#pragma unroll 1
      for (int v = 0; v < VOCAB; ++v) {
        float lg = lrow[v];
        gout[v] = lg;
        uint32_t h = wanghash((uint32_t)seed * 0x9E3779B9u ^
                              (uint32_t)(step + 1) * 0x85EBCA6Bu ^
                              (uint32_t)grow * 0xC2B2AE35u ^
                              (uint32_t)v * 0x27D4EB2Fu);
        float u = (float)(h >> 9) * (1.0f / 8388608.0f) + 1e-12f;
        float gum = -__logf(-__logf(u));
        float s = lg + gum;
        if (s > best) { best = s; bidx = v; }
      }
      samples_out[(size_t)grow * NC_STEPS + step] = bidx;
      const float* e = embed + (size_t)bidx * EMB;
      const int row = m0 + m;
      for (int c = 0; c < EMB; c += 4) {
        f32x4 v4 = *(const f32x4*)&e[c];
        *(bf16x4*)&xcat[row * XSTR + 256 + c] = cvt4(v4);
      }
    }

    // copy staged h_new into x_cat h columns (wave-private rows)
    {
      const int row = m0 + lo16;
      const int c0 = hi * 128;
      for (int c = c0; c < c0 + 128; c += 8) {
        bf16x8 v = *(const bf16x8*)&hstg[row * HSTR + c];
        *(bf16x8*)&xcat[row * XSTR + 320 + c] = v;
      }
    }
  }
}

// ---------------------------------------------------------------------------
// Launch
// ---------------------------------------------------------------------------
extern "C" void kernel_launch(void* const* d_in, const int* in_sizes, int n_in,
                              void* d_out, int out_size, void* d_ws,
                              size_t ws_size, hipStream_t stream) {
  const float* ctx         = (const float*)d_in[0];
  const float* embed_table = (const float*)d_in[1];
  const float* start_embed = (const float*)d_in[2];
  const float* Wp          = (const float*)d_in[3];
  const float* bp          = (const float*)d_in[4];
  const float* Wi          = (const float*)d_in[5];
  const float* Wh          = (const float*)d_in[6];
  const float* bh          = (const float*)d_in[7];
  const float* Wo          = (const float*)d_in[8];
  const float* bo          = (const float*)d_in[9];
  const int*   seed        = (const int*)d_in[10];

  char* ws = (char*)d_ws;
  __bf16* wcat = (__bf16*)(ws + WCAT_OFF);
  __bf16* wp   = (__bf16*)(ws + WP_OFF);
  __bf16* wo   = (__bf16*)(ws + WO_OFF);

  pack_wcat_kernel<<<(64 * KC_GATE * 512) / 256, 256, 0, stream>>>(Wi, Wh, wcat);
  pack_wp_kernel<<<(16 * KC_H * 512) / 256, 256, 0, stream>>>(Wp, wp);
  pack_wo_kernel<<<(2 * KC_H * 512) / 256, 256, 0, stream>>>(Wo, wo);

  float* logits_out = (float*)d_out;
  int* samples_out  = (int*)(logits_out + (size_t)BATCH * NC_STEPS * VOCAB);

  belief_decoder_main<<<BATCH / MTILE, 64, 0, stream>>>(
      ctx, embed_table, start_embed, bp, bh, bo, wcat, wp, wo, logits_out,
      samples_out, seed);
}
